// LSTMStep_81140522156907
// MI455X (gfx1250) — matrix operand order repaired and checked by reading, not compiled, not run on hardware
//
#include <hip/hip_runtime.h>
#include <hip/hip_bf16.h>
#include <math.h>

// Problem constants (from reference):
//   B=256, T=256, D=512, UNITS=1024, NCLASSES=512, 4*UNITS=4096
#define BB   256
#define TT   256
#define DD   512
#define UN   1024
#define G4   4096      // 4*UNITS
#define NC   512       // NCLASSES
#define KCAT 1536      // D + UNITS (fused [x_t | h] GEMM depth)

// CDNA5 direct global->LDS async copy (ASYNCcnt-tracked, no VGPR data path).
#define USE_ASYNC_LDS 1

typedef __attribute__((ext_vector_type(16))) __bf16 v16bf;
typedef __attribute__((ext_vector_type(8)))  float  v8f;

union FragAB { v16bf v; unsigned int d[8]; };
union FragC  { v8f  v; float f[8]; };
union Pack8  { uint4 q; unsigned short s[8]; };

__device__ __forceinline__ unsigned short f2bf(float f) {
  unsigned int u = __float_as_uint(f);
  unsigned int r = (u + 0x7FFFu + ((u >> 16) & 1u)) >> 16;  // RNE
  return (unsigned short)r;
}

__device__ __forceinline__ float hsig(float z) {
  return fminf(fmaxf(0.2f * z + 0.5f, 0.0f), 1.0f);
}

#if USE_ASYNC_LDS
// lds_off = low 32 bits of generic pointer to __shared__ (LDS byte offset).
// Per-lane 16B copy: LDS[lds_off] = MEM[gaddr].
#define ASYNC_B128_TO_LDS(lds_off, gaddr)                                  \
  asm volatile("global_load_async_to_lds_b128 %0, %1, off"                 \
               :: "v"(lds_off), "v"(gaddr) : "memory")
#define WAIT_ASYNCCNT0() asm volatile("s_wait_asynccnt 0x0" ::: "memory")
#endif

// ---------------------------------------------------------------------------
// Pack weights into WMMA B-fragment order (bf16).
// Logical matrix row k: k<split -> src0[k][n], else src1[k-split][n]; shape KxN.
// Fragment order used here (contiguous k, same as the A load): lane<16: N=lane, K=0..15;
// lane>=16: N=lane-16, K=16..31. Element e at K = 16*(lane>>4) + e.
// Flat: Bp[(((kt*(N/16)+nt)*32 + lane)*16 + e)]
// ---------------------------------------------------------------------------
__global__ void __launch_bounds__(256) pack_b_kernel(
    const float* __restrict__ src0, const float* __restrict__ src1, int split,
    unsigned short* __restrict__ Bp, int K, int N) {
  int id = blockIdx.x * 256 + threadIdx.x;         // covers K*N
  int e    = id & 15;
  int lane = (id >> 4) & 31;
  int rest = id >> 9;                              // kt*(N/16) + nt
  int ntiles = N >> 4;
  int nt = rest % ntiles;
  int kt = rest / ntiles;
  int k = kt * 32 + ((lane >> 4) << 4) + e;
  int n = nt * 16 + (lane & 15);
  float v = (k < split) ? src0[(size_t)k * N + n]
                        : src1[(size_t)(k - split) * N + n];
  Bp[id] = f2bf(v);
}

__global__ void __launch_bounds__(256) zero_state_kernel(
    float* __restrict__ c, unsigned short* __restrict__ H0) {
  int id = blockIdx.x * 256 + threadIdx.x;         // BB*UN
  c[id] = 0.0f;
  H0[id] = 0;
}

// ---------------------------------------------------------------------------
// Fully fused recurrent step:
//   z = [bf16(x_t) | h_{t-1}] @ [Wx;U]  + bias + W[512+idx]   (WMMA bf16)
//   c_t, h_t = LSTM cell (register-resident epilogue; zi/zf/zc/zo live in
//   the 4 accumulators because each wave owns the SAME u-range in all gates)
// Block: 256 thr = 8 waves; tile 64 rows x (32 u-cols x 4 gates).
// grid = (UN/32, BB/64) = (32, 4).
// ---------------------------------------------------------------------------
__global__ void __launch_bounds__(256) lstm_step_kernel(
    const float* __restrict__ x,              // [B,T,D] fp32
    const unsigned short* __restrict__ Hin,   // [B,UN] bf16  (h_{t-1})
    const unsigned short* __restrict__ Bp,    // WcatP packed (K=1536,N=4096)
    const float* __restrict__ bias,           // [4096]
    const float* __restrict__ W,              // [1024,4096] (word-row gather)
    const int* __restrict__ idx,              // [B] (prev argmax)
    float* __restrict__ c,                    // [B,UN] fp32
    unsigned short* __restrict__ Hout,        // [B,UN] bf16  (h_t)
    int t) {
  constexpr int LDA = 40;
  __shared__ __align__(16) unsigned short As[64 * LDA];

  const int tid  = threadIdx.x;
  const int lane = tid & 31;
  const int wid  = tid >> 5;
  const int wM   = wid & 3;                   // wave M slot (16 rows)
  const int wN   = wid >> 2;                  // wave u-half (16 u-cols)
  const int hlf  = lane >> 4;
  const int l16  = lane & 15;

  const int blockM = blockIdx.y * 64;
  const int u0     = blockIdx.x * 32 + wN * 16;   // this wave's u base
  const int row0   = blockM + wM * 16 + hlf * 8;  // C-frag row base

  // acc[g] = gate g (i,f,c,o) for cols u0 + (lane&15); init = bias (+ word row)
  FragC acc[4];
  #pragma unroll
  for (int r = 0; r < 8; ++r) {
    const int b = row0 + r;
    const int wrow = (t > 0) ? idx[b] : 0;
    #pragma unroll
    for (int g = 0; g < 4; ++g) {
      const int col = g * UN + u0 + l16;
      float v = bias[col];
      if (t > 0) v += W[(size_t)(DD + wrow) * G4 + col];
      acc[g].f[r] = v;
    }
  }

  const int ldrow = tid >> 2;                 // 0..63 (batch row in tile)
  const int ldcol = (tid & 3) * 8;            // 8 elements per thread

  for (int kb = 0; kb < (KCAT >> 5); ++kb) {
    __syncthreads();
    // Stage 64x32 A tile: K<512 -> convert fp32 x on the fly (VGPR path);
    // K>=512 -> bf16 h copied straight to LDS via async tensor path.
    {
      const int krow = blockM + ldrow;
      if (kb < (DD >> 5)) {
        const float* src = x + ((size_t)krow * TT + t) * DD + kb * 32 + ldcol;
        float4 f0 = *(const float4*)src;
        float4 f1 = *(const float4*)(src + 4);
        Pack8 pk;
        pk.s[0] = f2bf(f0.x); pk.s[1] = f2bf(f0.y);
        pk.s[2] = f2bf(f0.z); pk.s[3] = f2bf(f0.w);
        pk.s[4] = f2bf(f1.x); pk.s[5] = f2bf(f1.y);
        pk.s[6] = f2bf(f1.z); pk.s[7] = f2bf(f1.w);
        *(uint4*)&As[ldrow * LDA + ldcol] = pk.q;
        __builtin_prefetch(src + 32, 0, 1);     // next x tile
      } else {
        const unsigned short* src =
            Hin + (size_t)krow * UN + (kb * 32 - DD) + ldcol;
#if USE_ASYNC_LDS
        const unsigned int loff =
            (unsigned int)(size_t)&As[ldrow * LDA + ldcol];
        ASYNC_B128_TO_LDS(loff, src);
#else
        *(uint4*)&As[ldrow * LDA + ldcol] = *(const uint4*)src;
#endif
        __builtin_prefetch(src + 32, 0, 1);     // next h tile
      }
    }
#if USE_ASYNC_LDS
    WAIT_ASYNCCNT0();                           // drain this wave's async copies
#endif
    __syncthreads();

    // A fragment (contiguous k = 16*hlf + s, matching packed B)
    FragAB a;
    {
      const int arow = wM * 16 + l16;
      #pragma unroll
      for (int p = 0; p < 8; ++p) {
        const int kk = hlf * 16 + 2 * p;
        a.d[p] = *(const unsigned int*)&As[arow * LDA + kk];
      }
    }

    #pragma unroll
    for (int g = 0; g < 4; ++g) {
      const int ntg = g * (UN >> 4) + blockIdx.x * 2 + wN;  // N-tile of gate g
      const unsigned short* bsrc =
          Bp + ((((size_t)kb * (G4 >> 4) + ntg) * 32 + lane) << 4);
      FragAB bf;
      *(uint4*)&bf.d[0] = *(const uint4*)bsrc;
      *(uint4*)&bf.d[4] = *(const uint4*)(bsrc + 8);
      acc[g].v = __builtin_amdgcn_wmma_f32_16x16x32_bf16(
          false, a.v, false, bf.v, (short)0, acc[g].v, false, false);
    }
  }

  // Register-resident LSTM cell epilogue: acc = {zi, zf, zc, zo}
  #pragma unroll
  for (int r = 0; r < 8; ++r) {
    const int b = row0 + r;
    const int u = u0 + l16;
    const size_t ci = (size_t)b * UN + u;
    const float zi = acc[0].f[r], zf = acc[1].f[r];
    const float zc = acc[2].f[r], zo = acc[3].f[r];
    const float cn = hsig(zf) * c[ci] + hsig(zi) * tanhf(zc);
    c[ci] = cn;
    Hout[ci] = f2bf(hsig(zo) * tanhf(cn));
  }
}

// ---------------------------------------------------------------------------
// Fused output head: logits = h @ Wout + bout, then row argmax
// (softmax is monotone -> argmax of logits == argmax of probs;
//  jnp.argmax tie rule = first max).
// Block computes 16 rows x all 512 classes; 8 waves along N (64 cols each).
// grid = BB/16 = 16 blocks.
// ---------------------------------------------------------------------------
__global__ void __launch_bounds__(256) out_argmax_kernel(
    const unsigned short* __restrict__ H,     // [B,UN] bf16 (h_t)
    const unsigned short* __restrict__ Bp,    // WoutP packed (K=1024,N=512)
    const float* __restrict__ bout,           // [512]
    int* __restrict__ idxbuf,                 // [B]
    float* __restrict__ out,                  // [B,T] (as float)
    int t) {
  constexpr int LDA = 40;
  constexpr int LDL = NC + 16;                // padded logits row
  __shared__ __align__(16) unsigned short As[16 * LDA];
  __shared__ float Ls[16 * LDL];
  __shared__ float sval[256];
  __shared__ int   sidx[256];

  const int tid  = threadIdx.x;
  const int lane = tid & 31;
  const int wid  = tid >> 5;                  // 0..7 -> N offset 64*wid
  const int hlf  = lane >> 4;
  const int l16  = lane & 15;
  const int blockM = blockIdx.x * 16;

  FragC acc[4];
  #pragma unroll
  for (int nt = 0; nt < 4; ++nt)
    #pragma unroll
    for (int r = 0; r < 8; ++r) acc[nt].f[r] = 0.0f;

  for (int kb = 0; kb < (UN >> 5); ++kb) {
    __syncthreads();
    if (tid < 64) {                           // stage 16x32 A tile
      const int arow = tid >> 2, acol = (tid & 3) * 8;
      const unsigned short* src =
          H + (size_t)(blockM + arow) * UN + kb * 32 + acol;
#if USE_ASYNC_LDS
      const unsigned int loff = (unsigned int)(size_t)&As[arow * LDA + acol];
      ASYNC_B128_TO_LDS(loff, src);
#else
      *(uint4*)&As[arow * LDA + acol] = *(const uint4*)src;
#endif
      __builtin_prefetch(src + 32, 0, 1);
    }
#if USE_ASYNC_LDS
    WAIT_ASYNCCNT0();                         // outside the guard: all waves
#endif
    __syncthreads();

    FragAB a;
    #pragma unroll
    for (int p = 0; p < 8; ++p) {
      const int kk = hlf * 16 + 2 * p;
      a.d[p] = *(const unsigned int*)&As[l16 * LDA + kk];
    }

    #pragma unroll
    for (int nt = 0; nt < 4; ++nt) {
      const int ntg = wid * 4 + nt;
      const unsigned short* bsrc =
          Bp + ((((size_t)kb * (NC >> 4) + ntg) * 32 + lane) << 4);
      FragAB bf;
      *(uint4*)&bf.d[0] = *(const uint4*)bsrc;
      *(uint4*)&bf.d[4] = *(const uint4*)(bsrc + 8);
      acc[nt].v = __builtin_amdgcn_wmma_f32_16x16x32_bf16(
          false, a.v, false, bf.v, (short)0, acc[nt].v, false, false);
    }
  }

  // Spill logits (+bout) to LDS, then tie-correct argmax.
  #pragma unroll
  for (int nt = 0; nt < 4; ++nt) {
    const int col = wid * 64 + nt * 16 + l16;
    const float bo = bout[col];
    #pragma unroll
    for (int r = 0; r < 8; ++r)
      Ls[(8 * hlf + r) * LDL + col] = acc[nt].f[r] + bo;
  }
  __syncthreads();

  const int row  = tid >> 4;                  // 16 threads per row
  const int part = tid & 15;
  float best = -INFINITY; int bi = 0;
  for (int j = part; j < NC; j += 16) {       // ascending j -> first max
    const float v = Ls[row * LDL + j];
    if (v > best) { best = v; bi = j; }
  }
  sval[tid] = best; sidx[tid] = bi;
  __syncthreads();
  for (int s = 8; s > 0; s >>= 1) {
    if (part < s) {
      const float v2 = sval[tid + s]; const int i2 = sidx[tid + s];
      if (v2 > sval[tid] || (v2 == sval[tid] && i2 < sidx[tid])) {
        sval[tid] = v2; sidx[tid] = i2;
      }
    }
    __syncthreads();
  }
  if (part == 0) {
    const int b = blockM + row;
    idxbuf[b] = sidx[tid];
    out[(size_t)b * TT + t] = (float)sidx[tid];
  }
}

// ---------------------------------------------------------------------------
extern "C" void kernel_launch(void* const* d_in, const int* in_sizes, int n_in,
                              void* d_out, int out_size, void* d_ws, size_t ws_size,
                              hipStream_t stream) {
  const float* x    = (const float*)d_in[0];   // [B,T,D]
  const float* W    = (const float*)d_in[1];   // [D+NC, 4U]
  const float* U    = (const float*)d_in[2];   // [U, 4U]
  const float* bias = (const float*)d_in[3];   // [4U]
  const float* Wout = (const float*)d_in[4];   // [U, NC]
  const float* bout = (const float*)d_in[5];   // [NC]
  float* out = (float*)d_out;
  (void)in_sizes; (void)n_in; (void)out_size; (void)ws_size;

  // Workspace carve-up (~15.7 MB)
  char* p = (char*)d_ws;
  auto carve = [&](size_t bytes) -> void* {
    void* r = (void*)p; p += (bytes + 255) & ~(size_t)255; return r;
  };
  unsigned short* WcatP = (unsigned short*)carve((size_t)KCAT * G4 * 2);
  unsigned short* WoutP = (unsigned short*)carve((size_t)UN * NC * 2);
  unsigned short* Hb0   = (unsigned short*)carve((size_t)BB * UN * 2);
  unsigned short* Hb1   = (unsigned short*)carve((size_t)BB * UN * 2);
  float*          cbuf  = (float*)carve((size_t)BB * UN * 4);
  int*            idxbuf = (int*)carve((size_t)BB * 4);

  // One-time packs (weights constant across the sequence) + state zeroing
  pack_b_kernel<<<(KCAT * G4) / 256, 256, 0, stream>>>(W, U, DD, WcatP, KCAT, G4);
  pack_b_kernel<<<(UN * NC) / 256, 256, 0, stream>>>(Wout, Wout, UN, WoutP, UN, NC);
  zero_state_kernel<<<(BB * UN) / 256, 256, 0, stream>>>(cbuf, Hb0);

  const dim3 gridStep(UN / 32, BB / 64);   // 32 x 4 = 128 blocks
  const dim3 gridOut(BB / 16);             // 16 blocks

  for (int t = 0; t < TT; ++t) {
    unsigned short* hin  = (t & 1) ? Hb1 : Hb0;
    unsigned short* hout = (t & 1) ? Hb0 : Hb1;
    lstm_step_kernel<<<gridStep, 256, 0, stream>>>(
        x, hin, WcatP, bias, W, idxbuf, cbuf, hout, t);
    out_argmax_kernel<<<gridOut, 256, 0, stream>>>(
        hout, WoutP, bout, idxbuf, out, t);
  }
}
